// MLP_learner_68796786147776
// MI455X (gfx1250) — matrix operand order repaired and checked by reading, not compiled, not run on hardware
//
#include <hip/hip_runtime.h>
#include <hip/hip_bf16.h>
#include <cstddef>

// ---------- types ----------
typedef __attribute__((ext_vector_type(16))) __bf16          v16bf;
typedef __attribute__((ext_vector_type(8)))  float           v8f;
typedef __attribute__((ext_vector_type(8)))  unsigned short  ush8;
typedef __attribute__((ext_vector_type(4)))  unsigned short  ush4;

#define GBM 128
#define GBN 128
#define GBK 32
#define LDSS 40   // ushorts per LDS tile row = 80 bytes (16B-aligned, padded)

__device__ __forceinline__ unsigned short f32_to_bf16_rne(float f) {
  unsigned int u = __float_as_uint(f);
  u += 0x7FFFu + ((u >> 16) & 1u);          // round-to-nearest-even
  return (unsigned short)(u >> 16);
}

union FragU { ush8 u2[2]; v16bf v; };

// A fragment 16x32 bf16 (ISA 7.12.2): lane half h holds K chunks
// [8h .. 8h+7] (v0-3) and [16+8h .. 23+8h] (v4-7); lane&15 = M row.
__device__ __forceinline__ v16bf load_frag_a(const unsigned short* tile, int lane) {
  const int h = lane >> 4, r = lane & 15;
  const unsigned short* p = tile + r * LDSS;
  FragU f;
  f.u2[0] = *(const ush8*)(p + 8 * h);
  f.u2[1] = *(const ush8*)(p + 16 + 8 * h);
  return f.v;
}

// B fragment 32x16 bf16: lane&15 = N column, lane half h holds K chunks
// [8h .. 8h+7] (v0-3) and [16+8h .. 23+8h] (v4-7).
__device__ __forceinline__ v16bf load_frag_b(const unsigned short* tile, int lane) {
  const int h = lane >> 4, r = lane & 15;
  const unsigned short* p = tile + r * LDSS;
  FragU f;
  f.u2[0] = *(const ush8*)(p + 8 * h);
  f.u2[1] = *(const ush8*)(p + 16 + 8 * h);
  return f.v;
}

// ---------- CDNA5 async global->LDS copy (ASYNCcnt-tracked, no VGPR staging) ----
// GV mode: LDS address comes from the VDST VGPR; LDS offset = low 32 bits of the
// generic pointer to a __shared__ object (ISA 10.2 aperture rules).
__device__ __forceinline__ void g2l_async_b128(unsigned lds_off, const void* gaddr) {
  asm volatile("global_load_async_to_lds_b128 %0, %1, off"
               :: "v"(lds_off), "v"(gaddr) : "memory");
}
__device__ __forceinline__ void wait_async0() {
  asm volatile("s_wait_asynccnt 0x0" ::: "memory");
}

// ---------- f32 -> bf16 bulk convert ----------
__global__ __launch_bounds__(256)
void cvt_f32_to_bf16_vec(const float* __restrict__ in,
                         unsigned short* __restrict__ out, int n4) {
  int i = blockIdx.x * 256 + threadIdx.x;
  if (i < n4) {
    float4 v = ((const float4*)in)[i];
    ush4 u = { f32_to_bf16_rne(v.x), f32_to_bf16_rne(v.y),
               f32_to_bf16_rne(v.z), f32_to_bf16_rne(v.w) };
    ((ush4*)out)[i] = u;
  }
}

// ---------- unified bf16 NT GEMM: C[i,j] = act(sum_k A[i,k]*B[j,k] + bias[j]) --
// 128x128x32 block, 256 threads / 8 waves (2x4), 64x32 of C per wave
// -> 8 v_wmma_f32_16x16x32_bf16 per K-step per wave.
// Double-buffered LDS filled by global_load_async_to_lds_b128.
#define GF_RELU 1
#define GF_OBF16 2
__global__ __launch_bounds__(256)
void gemm_nt_bf16(const unsigned short* __restrict__ A,
                  const unsigned short* __restrict__ B,
                  const float* __restrict__ bias,
                  void* __restrict__ Cout,
                  int M, int N, int K, int flags) {
  __shared__ unsigned short As[2][GBM * LDSS];
  __shared__ unsigned short Bs[2][GBN * LDSS];
  const int i0 = blockIdx.y * GBM;
  const int j0 = blockIdx.x * GBN;
  const int tid  = threadIdx.x;
  const int lane = tid & 31;
  const int wave = tid >> 5;
  const int wr = wave >> 2, wc = wave & 3;   // 2x4 waves; wave owns 64x32 of C

  const v8f vzero = {0.f,0.f,0.f,0.f,0.f,0.f,0.f,0.f};
  v8f acc[4][2];
#pragma unroll
  for (int m = 0; m < 4; ++m) { acc[m][0] = vzero; acc[m][1] = vzero; }

  const int nk = K / GBK;

  // prologue: async-fill buffer 0 with K-tile 0 (2 x 16B chunks per thread/tile)
#pragma unroll
  for (int i = 0; i < 2; ++i) {
    int idx = tid + i * 256;                 // 0..511
    int row = idx >> 2, c8 = idx & 3;
    g2l_async_b128((unsigned)(size_t)&As[0][row * LDSS + c8 * 8],
                   A + (size_t)(i0 + row) * K + c8 * 8);
    g2l_async_b128((unsigned)(size_t)&Bs[0][row * LDSS + c8 * 8],
                   B + (size_t)(j0 + row) * K + c8 * 8);
  }

  for (int kt = 0; kt < nk; ++kt) {
    const int cur = kt & 1;
    wait_async0();                           // this wave's copies have landed
    __syncthreads();                         // all waves' copies visible

    if (kt + 1 < nk) {                       // fill other buffer during compute
      const int k0 = (kt + 1) * GBK, nxt = cur ^ 1;
#pragma unroll
      for (int i = 0; i < 2; ++i) {
        int idx = tid + i * 256;
        int row = idx >> 2, c8 = idx & 3;
        g2l_async_b128((unsigned)(size_t)&As[nxt][row * LDSS + c8 * 8],
                       A + (size_t)(i0 + row) * K + k0 + c8 * 8);
        g2l_async_b128((unsigned)(size_t)&Bs[nxt][row * LDSS + c8 * 8],
                       B + (size_t)(j0 + row) * K + k0 + c8 * 8);
      }
    }

    v16bf af[4], bfr[2];
#pragma unroll
    for (int m = 0; m < 4; ++m)
      af[m] = load_frag_a(&As[cur][(wr * 64 + m * 16) * LDSS], lane);
#pragma unroll
    for (int n = 0; n < 2; ++n)
      bfr[n] = load_frag_b(&Bs[cur][(wc * 32 + n * 16) * LDSS], lane);

#pragma unroll
    for (int m = 0; m < 4; ++m) {
      acc[m][0] = __builtin_amdgcn_wmma_f32_16x16x32_bf16(false, af[m], false, bfr[0], (short)0, acc[m][0], false, false);
      acc[m][1] = __builtin_amdgcn_wmma_f32_16x16x32_bf16(false, af[m], false, bfr[1], (short)0, acc[m][1], false, false);
    }
    // single barrier per iteration at loop head is sufficient: the compiler's
    // s_wait_dscnt before s_barrier_signal retires all frag reads of buf[cur]
    // before any wave can issue the kt+2 refill of it.
  }

  // C/D layout: VGPR r -> M = r + 8*(lane>=16), N = lane&15
  const int h = lane >> 4, n16 = lane & 15;
  const bool relu = (flags & GF_RELU) != 0;
  const bool obf  = (flags & GF_OBF16) != 0;
#pragma unroll
  for (int m = 0; m < 4; ++m)
#pragma unroll
    for (int n = 0; n < 2; ++n) {
      const int gn = j0 + wc * 32 + n * 16 + n16;
      const float bj = bias ? bias[gn] : 0.0f;
#pragma unroll
      for (int r = 0; r < 8; ++r) {
        const int gm = i0 + wr * 64 + m * 16 + 8 * h + r;
        float v = acc[m][n][r] + bj;
        if (relu) v = fmaxf(v, 0.0f);
        if (obf) ((unsigned short*)Cout)[(size_t)gm * N + gn] = f32_to_bf16_rne(v);
        else     ((float*)Cout)[(size_t)gm * N + gn] = v;
      }
    }
}

// ---------- row L2-normalize, emit bf16 ----------
__global__ __launch_bounds__(256)
void normalize_rows(const float* __restrict__ H, unsigned short* __restrict__ Hb, int D) {
  __shared__ float red[256];
  const int row = blockIdx.x;
  const float* hp = H + (size_t)row * D;
  float ss = 0.f;
  for (int i = threadIdx.x; i < D; i += 256) { float v = hp[i]; ss += v * v; }
  red[threadIdx.x] = ss;
  __syncthreads();
  for (int s = 128; s > 0; s >>= 1) {
    if (threadIdx.x < s) red[threadIdx.x] += red[threadIdx.x + s];
    __syncthreads();
  }
  const float scale = 1.0f / fmaxf(sqrtf(red[0]), 1e-12f);
  unsigned short* op = Hb + (size_t)row * D;
  for (int i = threadIdx.x; i < D; i += 256) op[i] = f32_to_bf16_rne(hp[i] * scale);
}

// ---------- per-row top-(k) threshold + relu mask, in place ----------
__global__ __launch_bounds__(256)
void topk_relu_rows(float* __restrict__ S, int N, int keep) {
  __shared__ float rowbuf[8192];   // 32 KB row staged in LDS (320 KB/WGP)
  __shared__ int   red[256];
  const int row = blockIdx.x;
  float* sp = S + (size_t)row * N;

  for (int i = threadIdx.x; i < N / 4; i += 256)
    ((float4*)rowbuf)[i] = ((const float4*)sp)[i];
  __syncthreads();

  // binary search the keep-th largest value; counts are monotone in mid
  float lo = -1.01f, hi = 1.01f;            // cosine sims live in [-1, 1]
  for (int it = 0; it < 25; ++it) {
    const float mid = 0.5f * (lo + hi);
    int c = 0;
    for (int i = threadIdx.x; i < N; i += 256) c += (rowbuf[i] >= mid) ? 1 : 0;
    red[threadIdx.x] = c;
    __syncthreads();
    for (int s = 128; s > 0; s >>= 1) {
      if (threadIdx.x < s) red[threadIdx.x] += red[threadIdx.x + s];
      __syncthreads();
    }
    const int cnt = red[0];
    __syncthreads();
    if (cnt >= keep) lo = mid; else hi = mid;
  }
  const float t = lo;                       // count(>= lo) >= keep
  for (int i = threadIdx.x; i < N; i += 256) {
    const float v = rowbuf[i];
    sp[i] = (v >= t) ? fmaxf(v, 0.0f) : 0.0f;
  }
}

// ---------- launch ----------
extern "C" void kernel_launch(void* const* d_in, const int* in_sizes, int n_in,
                              void* d_out, int out_size, void* d_ws, size_t ws_size,
                              hipStream_t stream) {
  (void)in_sizes; (void)n_in; (void)out_size; (void)ws_size;
  const int N = 8192, D = 512;
  const float* x = (const float*)d_in[0];
  const float* W = (const float*)d_in[1];   // [2, 512, 512]
  const float* b = (const float*)d_in[2];   // [2, 512]
  float* out = (float*)d_out;               // 8192 x 8192 f32

  char* ws = (char*)d_ws;
  unsigned short* Xb  = (unsigned short*)ws;                        // 8 MB @ 0
  unsigned short* h1b = (unsigned short*)(ws + ((size_t)8  << 20)); // 8 MB
  float*          h2  = (float*)        (ws + ((size_t)16 << 20));  // 16 MB
  unsigned short* Wb  = (unsigned short*)(ws + ((size_t)32 << 20)); // 1 MB
  unsigned short* Hb  = Xb;                 // overlays Xb once layer 1 is done

  // bulk f32 -> bf16 conversion of x and both weight layers
  cvt_f32_to_bf16_vec<<<dim3((N * D / 4 + 255) / 256), dim3(256), 0, stream>>>(x, Xb, N * D / 4);
  cvt_f32_to_bf16_vec<<<dim3((2 * D * D / 4 + 255) / 256), dim3(256), 0, stream>>>(W, Wb, 2 * D * D / 4);

  dim3 blk(256);
  // layer 1: relu, bf16 output
  gemm_nt_bf16<<<dim3(D / GBN, N / GBM), blk, 0, stream>>>(Xb, Wb, b, h1b, N, D, D, GF_RELU | GF_OBF16);
  // layer 2: linear, f32 output (feeds the norm)
  gemm_nt_bf16<<<dim3(D / GBN, N / GBM), blk, 0, stream>>>(h1b, Wb + (size_t)D * D, b + D, h2, N, D, D, 0);
  normalize_rows<<<dim3(N), dim3(256), 0, stream>>>(h2, Hb, D);
  // sim = Hb @ Hb^T, f32 output into d_out
  gemm_nt_bf16<<<dim3(N / GBN, N / GBM), blk, 0, stream>>>(Hb, Hb, nullptr, out, N, N, D, 0);
  topk_relu_rows<<<dim3(N), dim3(256), 0, stream>>>(out, N, 31);
}
